// Feature2VertexLayer_51256139710521
// MI455X (gfx1250) — matrix operand order repaired and checked by reading, not compiled, not run on hardware
//
#include <hip/hip_runtime.h>

// ---------------------------------------------------------------------------
// GCN-style net: per layer  out = deg_inv * (x@W0.T + b0 + segsum(x1 over edges))
// with x1 = x@W1.T + b1.   GEMMs via v_wmma_f32_16x16x32_bf16 with bf16x3
// error compensation; edge scatter via f32 global atomics (L2-resident).
// ---------------------------------------------------------------------------

typedef __bf16 bf16_t;
typedef bf16_t bf16x8 __attribute__((ext_vector_type(8)));
typedef bf16_t v16bf  __attribute__((ext_vector_type(16)));
typedef float  v8f    __attribute__((ext_vector_type(8)));

// ---- split fp32 -> (hi, lo) bf16 pair -------------------------------------
__global__ void split_f32_bf16x2(const float* __restrict__ src,
                                 bf16_t* __restrict__ hi,
                                 bf16_t* __restrict__ lo, long n) {
  long i = (long)blockIdx.x * blockDim.x + threadIdx.x;
  if (i >= n) return;
  float v = src[i];
  bf16_t h = (bf16_t)v;
  hi[i] = h;
  lo[i] = (bf16_t)(v - (float)h);
}

// ---- degree count (float) and reciprocal ----------------------------------
__global__ void deg_kernel(const int* __restrict__ edges, float* __restrict__ deg, int E) {
  int j = blockIdx.x * blockDim.x + threadIdx.x;
  if (j >= E) return;
  atomicAdd(deg + edges[2 * j], 1.0f);
  atomicAdd(deg + edges[2 * j + 1], 1.0f);
}

__global__ void recip_kernel(float* __restrict__ deg, int V) {
  int i = blockIdx.x * blockDim.x + threadIdx.x;
  if (i >= V) return;
  deg[i] = 1.0f / deg[i];
}

// ---- fused double GEMM: Y(V x 2*dout) = X(V x din) @ Wcat.T + [b0;b1] -----
// Wcat rows [0,dout) = W0, rows [dout,2*dout) = W1, each row length din.
// bf16x3: acc += Ah*Bh + Ah*Bl + Al*Bh  (fp32 accumulate).
__global__ __launch_bounds__(256) void gemm_bf16x3_kernel(
    const bf16_t* __restrict__ Xh, const bf16_t* __restrict__ Xl,
    const bf16_t* __restrict__ Wh, const bf16_t* __restrict__ Wl,
    const float* __restrict__ b0, const float* __restrict__ b1,
    float* __restrict__ Y, int V, int din, int dout) {
  const int N = 2 * dout;
  const int ntiles = (N + 15) >> 4;
  const int mtiles = V >> 4;  // V divisible by 16 (100000 = 6250*16)
  const int total = mtiles * ntiles;

  int wave = (int)((blockIdx.x * blockDim.x + threadIdx.x) >> 5);
  if (wave >= total) return;
  int mt = wave / ntiles;
  int nt = wave - mt * ntiles;

  int lane  = threadIdx.x & 31;
  int half  = lane >> 4;   // 0: lanes 0-15, 1: lanes 16-31
  int sub   = lane & 15;

  int arow = mt * 16 + sub;          // A: row index (M)
  int ncol = nt * 16 + sub;          // B/C/D: column index (N)
  int nclamp = ncol < N ? ncol : N - 1;

  const bf16_t* ah_p = Xh + (size_t)arow * din;
  const bf16_t* al_p = Xl + (size_t)arow * din;
  const bf16_t* bh_p = Wh + (size_t)nclamp * din;
  const bf16_t* bl_p = Wl + (size_t)nclamp * din;

  v8f acc = {};
  for (int k = 0; k < din; k += 32) {
    union U { v16bf v; bf16x8 h[2]; } ah, al, bh, bl;
    // A 16x32 bf16 layout: lane half selects K sub-chunk {8*half, 16+8*half}
    ah.h[0] = *(const bf16x8*)(ah_p + k + 8 * half);
    ah.h[1] = *(const bf16x8*)(ah_p + k + 16 + 8 * half);
    al.h[0] = *(const bf16x8*)(al_p + k + 8 * half);
    al.h[1] = *(const bf16x8*)(al_p + k + 16 + 8 * half);
    // B 32x16 bf16 layout: lane half selects K sub-chunk {8*half, 16+8*half}
    bh.h[0] = *(const bf16x8*)(bh_p + k + 8 * half);
    bh.h[1] = *(const bf16x8*)(bh_p + k + 16 + 8 * half);
    bl.h[0] = *(const bf16x8*)(bl_p + k + 8 * half);
    bl.h[1] = *(const bf16x8*)(bl_p + k + 16 + 8 * half);

    acc = __builtin_amdgcn_wmma_f32_16x16x32_bf16(false, ah.v, false, bh.v,
                                                  (short)0, acc, false, false);
    acc = __builtin_amdgcn_wmma_f32_16x16x32_bf16(false, ah.v, false, bl.v,
                                                  (short)0, acc, false, false);
    acc = __builtin_amdgcn_wmma_f32_16x16x32_bf16(false, al.v, false, bh.v,
                                                  (short)0, acc, false, false);
  }

  if (ncol < N) {
    float bias = (ncol < dout) ? b0[ncol] : b1[ncol - dout];
#pragma unroll
    for (int r = 0; r < 8; ++r) {
      int row = mt * 16 + r + 8 * half;  // C/D: VGPR r, lanes>=16 shifted by 8
      Y[(size_t)row * N + ncol] = acc[r] + bias;
    }
  }
}

// ---- edge scatter: nbr[e0] += y1[e1]; nbr[e1] += y1[e0] -------------------
// one wave per (edge, direction); y1 = Y[:, dout:2*dout]
__global__ void scatter_kernel(const int* __restrict__ edges,
                               const float* __restrict__ Y,
                               float* __restrict__ nbr, int E, int dout) {
  int wave = (int)((blockIdx.x * blockDim.x + threadIdx.x) >> 5);
  int lane = threadIdx.x & 31;
  if (wave >= 2 * E) return;
  int j = wave >> 1;
  int dir = wave & 1;
  int e0 = edges[2 * j];
  int e1 = edges[2 * j + 1];
  int dst = dir ? e1 : e0;
  int src = dir ? e0 : e1;
  const float* yr = Y + (size_t)src * (2 * dout) + dout;
  float* nr = nbr + (size_t)dst * dout;
  for (int c = lane; c < dout; c += 32) atomicAdd(nr + c, yr[c]);
}

// ---- combine: val = deg_inv * (y0 + nbr), optional relu; write next x -----
__global__ void combine_kernel(const float* __restrict__ Y,
                               const float* __restrict__ nbr,
                               const float* __restrict__ deginv,
                               bf16_t* __restrict__ Xh, bf16_t* __restrict__ Xl,
                               float* __restrict__ out, int V, int dout, int relu) {
  long i = (long)blockIdx.x * blockDim.x + threadIdx.x;
  long total = (long)V * dout;
  if (i >= total) return;
  int v = (int)(i / dout);
  int c = (int)(i - (long)v * dout);
  float y0 = Y[(size_t)v * (2 * dout) + c];
  float val = deginv[v] * (y0 + nbr[i]);
  if (relu) val = fmaxf(val, 0.0f);
  if (out) {
    out[i] = val;
  } else {
    bf16_t h = (bf16_t)val;
    Xh[i] = h;
    Xl[i] = (bf16_t)(val - (float)h);
  }
}

// ---------------------------------------------------------------------------
static inline size_t ws_take(size_t& off, size_t bytes) {
  size_t o = off;
  off = (off + bytes + 255) & ~(size_t)255;
  return o;
}

extern "C" void kernel_launch(void* const* d_in, const int* in_sizes, int n_in,
                              void* d_out, int out_size, void* d_ws, size_t ws_size,
                              hipStream_t stream) {
  const int DIMS[5] = {256, 192, 128, 64, 3};
  const int NL = 4;
  const int V = in_sizes[0] / 256;  // 100000
  const int E = in_sizes[1] / 2;    // 400000

  const float* features = (const float*)d_in[0];
  const int* edges = (const int*)d_in[1];
  const float* W0p[4] = {(const float*)d_in[2], (const float*)d_in[3],
                         (const float*)d_in[4], (const float*)d_in[5]};
  const float* b0p[4] = {(const float*)d_in[6], (const float*)d_in[7],
                         (const float*)d_in[8], (const float*)d_in[9]};
  const float* W1p[4] = {(const float*)d_in[10], (const float*)d_in[11],
                         (const float*)d_in[12], (const float*)d_in[13]};
  const float* b1p[4] = {(const float*)d_in[14], (const float*)d_in[15],
                         (const float*)d_in[16], (const float*)d_in[17]};

  // workspace carve-up (peak ~335 MB)
  char* ws = (char*)d_ws;
  size_t off = 0;
  float*  deg = (float*)(ws + ws_take(off, (size_t)V * 4));
  bf16_t* Xh  = (bf16_t*)(ws + ws_take(off, (size_t)V * 256 * 2));
  bf16_t* Xl  = (bf16_t*)(ws + ws_take(off, (size_t)V * 256 * 2));
  float*  Y   = (float*)(ws + ws_take(off, (size_t)V * 384 * 4));
  float*  nbr = (float*)(ws + ws_take(off, (size_t)V * 192 * 4));
  bf16_t* Wh  = (bf16_t*)(ws + ws_take(off, (size_t)384 * 256 * 2));
  bf16_t* Wl  = (bf16_t*)(ws + ws_take(off, (size_t)384 * 256 * 2));
  (void)ws_size;

  // degree -> deg_inv (recomputed every call; deterministic)
  hipMemsetAsync(deg, 0, (size_t)V * 4, stream);
  {
    int thr = 256, blk = (E + thr - 1) / thr;
    deg_kernel<<<blk, thr, 0, stream>>>(edges, deg, E);
    blk = (V + thr - 1) / thr;
    recip_kernel<<<blk, thr, 0, stream>>>(deg, V);
  }

  // x -> (Xh, Xl)
  {
    long n = (long)V * 256;
    int thr = 256;
    long blk = (n + thr - 1) / thr;
    split_f32_bf16x2<<<(int)blk, thr, 0, stream>>>(features, Xh, Xl, n);
  }

  for (int li = 0; li < NL; ++li) {
    int din = DIMS[li], dout = DIMS[li + 1];
    int N = 2 * dout;

    // weights -> split bf16, concatenated [W0; W1]
    {
      long n = (long)dout * din;
      int thr = 256;
      long blk = (n + thr - 1) / thr;
      split_f32_bf16x2<<<(int)blk, thr, 0, stream>>>(W0p[li], Wh, Wl, n);
      split_f32_bf16x2<<<(int)blk, thr, 0, stream>>>(W1p[li], Wh + n, Wl + n, n);
    }

    // Y = X @ Wcat.T + [b0;b1]
    {
      int mtiles = V / 16;
      int ntiles = (N + 15) / 16;
      long waves = (long)mtiles * ntiles;
      long threads = waves * 32;
      int thr = 256;
      long blk = (threads + thr - 1) / thr;
      gemm_bf16x3_kernel<<<(int)blk, thr, 0, stream>>>(Xh, Xl, Wh, Wl, b0p[li],
                                                       b1p[li], Y, V, din, dout);
    }

    // nbr = segment_sum of y1 rows over both edge directions
    hipMemsetAsync(nbr, 0, (size_t)V * dout * 4, stream);
    {
      long waves = 2L * E;
      long threads = waves * 32;
      int thr = 256;
      long blk = (threads + thr - 1) / thr;
      scatter_kernel<<<(int)blk, thr, 0, stream>>>(edges, Y, nbr, E, dout);
    }

    // combine (+relu, + re-split to bf16 for next layer / fp32 final out)
    {
      long n = (long)V * dout;
      int thr = 256;
      long blk = (n + thr - 1) / thr;
      int last = (li == NL - 1);
      combine_kernel<<<(int)blk, thr, 0, stream>>>(
          Y, nbr, deg, Xh, Xl, last ? (float*)d_out : nullptr, V, dout,
          last ? 0 : 1);
    }
  }
  (void)n_in;
  (void)out_size;
}
